// GNN_node_85624468013522
// MI455X (gfx1250) — matrix operand order repaired and checked by reading, not compiled, not run on hardware
//
#include <hip/hip_runtime.h>

#define D 128
#define NL 5

typedef __attribute__((ext_vector_type(16))) __bf16 v16bf;
typedef __attribute__((ext_vector_type(8)))  float  v8f;

__device__ __forceinline__ void atomAddF(float* p, float v) {
    unsafeAtomicAdd(p, v);   // native global_atomic_add_f32 path
}

// ---------------- zero (float4-vectorized) ----------------
__global__ void zero4_k(float4* __restrict__ p, int n4) {
    int i = blockIdx.x * blockDim.x + threadIdx.x;
    if (i < n4) p[i] = float4{0.f, 0.f, 0.f, 0.f};
}

// ---------------- h0 = atom_emb[x_ids] ----------------
__global__ void gather_k(const float* __restrict__ emb, const int* __restrict__ ids,
                         float* __restrict__ h, int N) {
    int i = blockIdx.x * blockDim.x + threadIdx.x;
    if (i >= N * 32) return;
    int node = i >> 5, c = i & 31;
    ((float4*)h)[i] = ((const float4*)emb)[(size_t)ids[node] * 32 + c];
}

// ---------------- repack fp32 weights -> bf16 WMMA B-fragment order ----------------
// frag(ct,ks): lane L holds elements i=0..15 with k = ks*32 + (L>>4)*8 + (i&7) + (i>>3)*16, n = ct*16 + (L&15)
__global__ void repack_k(const float* __restrict__ W1, const float* __restrict__ W2,
                         unsigned short* __restrict__ Wp) {
    int t = blockIdx.x * blockDim.x + threadIdx.x;
    if (t >= 10 * 16384) return;
    int i    = t & 15;
    int lane = (t >> 4) & 31;
    int ks   = (t >> 9) & 3;
    int ct   = (t >> 11) & 7;
    int m    = t >> 14;
    int k = ks * 32 + (lane >> 4) * 8 + (i & 7) + (i >> 3) * 16;
    int n = ct * 16 + (lane & 15);
    const float* Wb = (m < NL) ? (W1 + (size_t)m * D * D) : (W2 + (size_t)(m - NL) * D * D);
    ((__bf16*)Wp)[t] = (__bf16)Wb[(size_t)k * D + n];
}

// ---------------- edge scatter: agg[dst] += relu(h[src] + bond[attr]) ----------------
__global__ void msgpass_k(const float* __restrict__ h, const float* __restrict__ bondl,
                          const int* __restrict__ src, const int* __restrict__ dst,
                          const int* __restrict__ attr, float* __restrict__ agg, int E) {
    int t = blockIdx.x * blockDim.x + threadIdx.x;
    int e = t >> 5;
    if (e >= E) return;
    int lane = t & 31;                              // e is wave-uniform -> scalar idx loads
    int s = src[e], d = dst[e], a = attr[e];
    float4 hv = ((const float4*)(h + (size_t)s * D))[lane];
    float4 bv = ((const float4*)(bondl + (size_t)a * D))[lane];
    float* ap = agg + (size_t)d * D + lane * 4;
    atomAddF(ap + 0, fmaxf(hv.x + bv.x, 0.f));
    atomAddF(ap + 1, fmaxf(hv.y + bv.y, 0.f));
    atomAddF(ap + 2, fmaxf(hv.z + bv.z, 0.f));
    atomAddF(ap + 3, fmaxf(hv.w + bv.w, 0.f));
}

// ---------------- fused WMMA GEMM: Y = f(A) @ W + bias, + column stats ----------------
// mode 0: a = (1+eps[l])*A0 + A1        (GIN combine, pre-MLP)
// mode 1: a = relu(A0*scale[k]+shift[k]) (BN+ReLU fused into operand load)
__global__ __launch_bounds__(128) void gemm128_wmma(
    const float* __restrict__ A0, const float* __restrict__ A1,
    const float* __restrict__ scale, const float* __restrict__ shift,
    const float* __restrict__ epsArr, int layer, int mode,
    const unsigned short* __restrict__ Wp, const float* __restrict__ bias,
    float* __restrict__ Y, float* __restrict__ colsum,
    float* __restrict__ colsumsq, int N) {
    const int tid  = threadIdx.x;
    const int wave = tid >> 5;
    const int lane = tid & 31;
    const int lo   = lane & 15;
    const int hi   = lane >> 4;
    const int rowbase = blockIdx.x * 64 + wave * 16;
    int arow = rowbase + lo;
    if (arow >= N) arow = N - 1;
    const int kb = hi * 8;
    const float c0 = (mode == 0) ? (1.0f + epsArr[layer]) : 0.0f;

    const v16bf* __restrict__ Bf = (const v16bf*)Wp;
    const float* __restrict__ A0r = A0 + (size_t)arow * D;
    const float* __restrict__ A1r = (mode == 0) ? (A1 + (size_t)arow * D) : A0;

    v8f acc[8] = {};

#pragma unroll
    for (int ks = 0; ks < 4; ++ks) {
        const int kA = ks * 32 + kb;                // 32B-aligned
        float pa[16], va[16];
        *(float4*)(pa + 0)  = *(const float4*)(A0r + kA);
        *(float4*)(pa + 4)  = *(const float4*)(A0r + kA + 4);
        *(float4*)(pa + 8)  = *(const float4*)(A0r + kA + 16);
        *(float4*)(pa + 12) = *(const float4*)(A0r + kA + 20);
        if (mode == 0) {
            float qa[16];
            *(float4*)(qa + 0)  = *(const float4*)(A1r + kA);
            *(float4*)(qa + 4)  = *(const float4*)(A1r + kA + 4);
            *(float4*)(qa + 8)  = *(const float4*)(A1r + kA + 16);
            *(float4*)(qa + 12) = *(const float4*)(A1r + kA + 20);
#pragma unroll
            for (int i = 0; i < 16; ++i) va[i] = fmaf(c0, pa[i], qa[i]);
        } else {
            float sa[16], ta[16];
            *(float4*)(sa + 0)  = *(const float4*)(scale + kA);
            *(float4*)(sa + 4)  = *(const float4*)(scale + kA + 4);
            *(float4*)(sa + 8)  = *(const float4*)(scale + kA + 16);
            *(float4*)(sa + 12) = *(const float4*)(scale + kA + 20);
            *(float4*)(ta + 0)  = *(const float4*)(shift + kA);
            *(float4*)(ta + 4)  = *(const float4*)(shift + kA + 4);
            *(float4*)(ta + 8)  = *(const float4*)(shift + kA + 16);
            *(float4*)(ta + 12) = *(const float4*)(shift + kA + 20);
#pragma unroll
            for (int i = 0; i < 16; ++i) va[i] = fmaxf(fmaf(pa[i], sa[i], ta[i]), 0.f);
        }
        v16bf a;
#pragma unroll
        for (int i = 0; i < 16; ++i) a[i] = (__bf16)va[i];
#pragma unroll
        for (int ct = 0; ct < 8; ++ct) {
            v16bf b = Bf[(ct * 4 + ks) * 32 + lane];
            acc[ct] = __builtin_amdgcn_wmma_f32_16x16x32_bf16(
                false, a, false, b, (short)0, acc[ct], false, false);
        }
    }

    // epilogue: bias add, store, per-column sum / sumsq via xor-16 reduce + atomics
#pragma unroll
    for (int ct = 0; ct < 8; ++ct) {
        const int n = ct * 16 + lo;
        const float bv = bias[n];
        float s = 0.f, q = 0.f;
#pragma unroll
        for (int v = 0; v < 8; ++v) {
            const int row = rowbase + v + 8 * hi;   // C/D layout: VGPR v -> M = v + 8*hi
            float o = acc[ct][v] + bv;
            if (row < N) {
                Y[(size_t)row * D + n] = o;
                s += o;
                q += o * o;
            }
        }
        s += __shfl_xor(s, 16, 32);
        q += __shfl_xor(q, 16, 32);
        if (hi == 0) {
            atomAddF(&colsum[n], s);
            atomAddF(&colsumsq[n], q);
        }
    }
}

// ---------------- BN prep: scale/shift from column sums ----------------
__global__ void bnprep_k(const float* __restrict__ cs, const float* __restrict__ cq,
                         const float* __restrict__ g, const float* __restrict__ bt,
                         float* __restrict__ scale, float* __restrict__ shift, int N) {
    int k = threadIdx.x;
    float inv = 1.0f / (float)N;
    float m = cs[k] * inv;
    float v = fmaxf(cq[k] * inv - m * m, 0.f);
    float sc = g[k] * rsqrtf(v + 1e-5f);
    scale[k] = sc;
    shift[k] = bt[k] - m * sc;
}

// ---------------- residual: h += (relu?)(y2*scale+shift) ----------------
__global__ void finalize_k(float* __restrict__ h, const float* __restrict__ y2,
                           const float* __restrict__ scale, const float* __restrict__ shift,
                           int relu, int n4) {
    int i = blockIdx.x * blockDim.x + threadIdx.x;
    if (i >= n4) return;
    int k4 = i & 31;
    float4 y  = ((const float4*)y2)[i];
    float4 sc = ((const float4*)scale)[k4];
    float4 sh = ((const float4*)shift)[k4];
    float4 hv = ((float4*)h)[i];
    float4 z;
    z.x = fmaf(y.x, sc.x, sh.x);
    z.y = fmaf(y.y, sc.y, sh.y);
    z.z = fmaf(y.z, sc.z, sh.z);
    z.w = fmaf(y.w, sc.w, sh.w);
    if (relu) {
        z.x = fmaxf(z.x, 0.f); z.y = fmaxf(z.y, 0.f);
        z.z = fmaxf(z.z, 0.f); z.w = fmaxf(z.w, 0.f);
    }
    hv.x += z.x; hv.y += z.y; hv.z += z.z; hv.w += z.w;
    ((float4*)h)[i] = hv;
}

extern "C" void kernel_launch(void* const* d_in, const int* in_sizes, int n_in,
                              void* d_out, int out_size, void* d_ws, size_t ws_size,
                              hipStream_t stream) {
    const int N = in_sizes[0];
    const int E = in_sizes[2];

    const int*   x_ids    = (const int*)d_in[0];
    const int*   ei       = (const int*)d_in[1];   // [2,E] row-major: src=ei, dst=ei+E
    const int*   eattr    = (const int*)d_in[2];
    const float* atom_emb = (const float*)d_in[3];
    const float* bond_emb = (const float*)d_in[4];
    const float* W1       = (const float*)d_in[5];
    const float* b1       = (const float*)d_in[6];
    const float* g1       = (const float*)d_in[7];
    const float* bt1      = (const float*)d_in[8];
    const float* W2       = (const float*)d_in[9];
    const float* b2       = (const float*)d_in[10];
    const float* eps      = (const float*)d_in[11];
    const float* g_out    = (const float*)d_in[12];
    const float* b_out    = (const float*)d_in[13];
    float* h = (float*)d_out;

    // workspace carve: agg[N*D] (reused as y2) | y1[N*D] | packed weights | stats
    float* agg = (float*)d_ws;
    float* y1  = agg + (size_t)N * D;
    unsigned short* Wp = (unsigned short*)(y1 + (size_t)N * D);
    float* stats = (float*)((char*)Wp + (size_t)10 * 16384 * sizeof(unsigned short));
    float* csA = stats;       float* cqA = stats + 128;
    float* scA = stats + 256; float* shA = stats + 384;
    float* csB = stats + 512; float* cqB = stats + 640;
    float* scB = stats + 768; float* shB = stats + 896;
    float* y2  = agg;  // gemm2 never reads agg -> safe alias

    repack_k<<<(10 * 16384 + 255) / 256, 256, 0, stream>>>(W1, W2, Wp);
    gather_k<<<(N * 32 + 255) / 256, 256, 0, stream>>>(atom_emb, x_ids, h, N);

    for (int l = 0; l < NL; ++l) {
        zero4_k<<<(N * 32 + 255) / 256, 256, 0, stream>>>((float4*)agg, N * 32);
        zero4_k<<<1, 256, 0, stream>>>((float4*)stats, 256);
        msgpass_k<<<(E * 32 + 255) / 256, 256, 0, stream>>>(
            h, bond_emb + (size_t)l * 4 * D, ei, ei + E, eattr, agg, E);
        gemm128_wmma<<<(N + 63) / 64, 128, 0, stream>>>(
            h, agg, scA, shA, eps, l, 0, Wp + (size_t)l * 16384, b1 + (size_t)l * D,
            y1, csA, cqA, N);
        bnprep_k<<<1, 128, 0, stream>>>(csA, cqA, g1 + (size_t)l * D, bt1 + (size_t)l * D,
                                        scA, shA, N);
        gemm128_wmma<<<(N + 63) / 64, 128, 0, stream>>>(
            y1, y1, scA, shA, eps, l, 1, Wp + (size_t)(NL + l) * 16384, b2 + (size_t)l * D,
            y2, csB, cqB, N);
        bnprep_k<<<1, 128, 0, stream>>>(csB, cqB, g_out + (size_t)l * D, b_out + (size_t)l * D,
                                        scB, shB, N);
        finalize_k<<<(N * 32 + 255) / 256, 256, 0, stream>>>(h, y2, scB, shB,
                                                             (l < NL - 1) ? 1 : 0, N * 32);
    }
}
